// CrossAttention_30520037605769
// MI455X (gfx1250) — matrix operand order repaired and checked by reading, not compiled, not run on hardware
//
#include <hip/hip_runtime.h>

// ---------- types ----------
typedef __bf16 bf16;
typedef __attribute__((ext_vector_type(16))) __bf16 v16bf;
typedef __attribute__((ext_vector_type(8)))  __bf16 v8bf;
typedef __attribute__((ext_vector_type(8)))  float  v8f;
typedef __attribute__((ext_vector_type(4)))  float  v4f;

__device__ __forceinline__ bf16 to_bf(float x)  { return (bf16)x; }
__device__ __forceinline__ bf16 to_bf(bf16 x)   { return x; }

__device__ __forceinline__ v16bf frag16(v8bf lo, v8bf hi) {
    return __builtin_shufflevector(lo, hi, 0,1,2,3,4,5,6,7,8,9,10,11,12,13,14,15);
}

__device__ __forceinline__ v8f wmma_bf16(v16bf a, v16bf b, v8f c) {
    return __builtin_amdgcn_wmma_f32_16x16x32_bf16(
        /*neg_a=*/false, a, /*neg_b=*/false, b,
        /*c_mod=*/(short)0, c, /*reuse_a=*/false, /*reuse_b=*/false);
}

// ==========================================================================
// GEMM with bias: C[M,N] = A[M,K] * W[K,N] + bias[N]
// A: f32 or bf16 (row-major). W,bias: f32. C: f32 or bf16.
// Block tile 128(M) x 64(N), K-step 32, 8 waves, wave tile 32x32 (2x2 WMMA).
// Double-buffered LDS: one __syncthreads per K-step, global loads for the
// next tile are issued before the compute on the current tile.
// ==========================================================================
#define GM_BM 128
#define GM_BN 64
#define GM_BK 32
#define GM_LD (GM_BK + 8)   // padded LDS row (keeps 16B chunk alignment)

template <typename TIN, typename TOUT>
__global__ __launch_bounds__(256)
void gemm_bias_wmma(const TIN* __restrict__ A, const float* __restrict__ W,
                    const float* __restrict__ bias, TOUT* __restrict__ C,
                    int M, int N, int K)
{
    __shared__ __align__(16) bf16 As[2][GM_BM][GM_LD];
    __shared__ __align__(16) bf16 Wt[2][GM_BN][GM_LD];  // W tile transposed [n][k]

    const int tid  = threadIdx.x;
    const int lane = tid & 31;
    const int wid  = tid >> 5;
    const int bm   = blockIdx.y * GM_BM;
    const int bn   = blockIdx.x * GM_BN;

    const int wm = (wid >> 1) * 32;     // wave M offset in block
    const int wn = (wid & 1)  * 32;     // wave N offset in block

    v8f acc[2][2] = {};

    // loader mapping
    const int lrow = tid >> 1;           // 0..127  (A tile row)
    const int lcol = (tid & 1) * 16;     // 0|16    (A tile col base, 16 elems)
    const int wk   = tid >> 3;           // 0..31   (W tile k)
    const int wn0  = (tid & 7) * 8;      // 0..56   (W tile n base, 8 elems)

    // fragment lane mapping (contiguous k = 16*half + s for A and B)
    const int r     = lane & 15;
    const int half  = lane >> 4;         // 0|1
    const int halfA = half * 8;          // unused; A uses halfB too
    const int halfB = half * 16;         // B: K-half select

    // staging registers
    v4f  aF[4]; v8bf aB[2]; v4f wR[2];

    auto gload = [&](int k0) {
        const TIN* src = A + (size_t)(bm + lrow) * K + k0 + lcol;
        if constexpr (sizeof(TIN) == 4) {
            const v4f* s4 = (const v4f*)src;
            #pragma unroll
            for (int i = 0; i < 4; ++i) aF[i] = s4[i];
        } else {
            const v8bf* s8 = (const v8bf*)src;
            #pragma unroll
            for (int i = 0; i < 2; ++i) aB[i] = s8[i];
        }
        const v4f* w4 = (const v4f*)(W + (size_t)(k0 + wk) * N + bn + wn0);
        wR[0] = w4[0]; wR[1] = w4[1];
    };
    auto sstore = [&](int buf) {
        if constexpr (sizeof(TIN) == 4) {
            #pragma unroll
            for (int i = 0; i < 4; ++i)
                #pragma unroll
                for (int j = 0; j < 4; ++j) As[buf][lrow][lcol + i*4 + j] = to_bf(aF[i][j]);
        } else {
            #pragma unroll
            for (int i = 0; i < 2; ++i)
                #pragma unroll
                for (int j = 0; j < 8; ++j) As[buf][lrow][lcol + i*8 + j] = aB[i][j];
        }
        #pragma unroll
        for (int i = 0; i < 2; ++i)
            #pragma unroll
            for (int j = 0; j < 4; ++j) Wt[buf][wn0 + i*4 + j][wk] = to_bf(wR[i][j]);
    };

    gload(0);
    sstore(0);
    __syncthreads();

    int cur = 0;
    for (int k0 = 0; k0 < K; k0 += GM_BK) {
        const bool hasNext = (k0 + GM_BK) < K;
        if (hasNext) gload(k0 + GM_BK);      // issue next-tile global loads early

        // --- build fragments and multiply from current buffer ---
        v16bf af[2], bfr[2];
        #pragma unroll
        for (int mi = 0; mi < 2; ++mi) {
            const bf16* p = &As[cur][wm + mi*16 + r][0];
            af[mi] = frag16(*(const v8bf*)(p + halfB), *(const v8bf*)(p + halfB + 8));
        }
        #pragma unroll
        for (int ni = 0; ni < 2; ++ni) {
            const bf16* p = &Wt[cur][wn + ni*16 + r][halfB];
            bfr[ni] = frag16(*(const v8bf*)p, *(const v8bf*)(p + 8));
        }
        #pragma unroll
        for (int mi = 0; mi < 2; ++mi)
            #pragma unroll
            for (int ni = 0; ni < 2; ++ni)
                acc[mi][ni] = wmma_bf16(af[mi], bfr[ni], acc[mi][ni]);

        if (hasNext) sstore(cur ^ 1);        // write next tile to alternate buffer
        __syncthreads();
        cur ^= 1;
    }

    // --- epilogue: bias add + store ---
    #pragma unroll
    for (int mi = 0; mi < 2; ++mi) {
        #pragma unroll
        for (int ni = 0; ni < 2; ++ni) {
            const int col = bn + wn + ni*16 + r;
            const float bcol = bias[col];
            #pragma unroll
            for (int j = 0; j < 8; ++j) {
                const int row = bm + wm + mi*16 + half*8 + j;
                float v = acc[mi][ni][j] + bcol;
                C[(size_t)row * N + col] = (TOUT)v;
            }
        }
    }
}

// ==========================================================================
// V transpose: KVbf[b, z, F + h*64 + d]  ->  Vt[((b*H+h)*64 + d), z]   (bf16)
// ==========================================================================
__global__ __launch_bounds__(256)
void v_transpose_kernel(const bf16* __restrict__ kv, bf16* __restrict__ vt,
                        int Z, int F, int H)
{
    int idx = blockIdx.x * blockDim.x + threadIdx.x;   // over B*Z*F
    int c   = idx % F;
    int row = idx / F;          // b*Z + z
    int z   = row % Z;
    int b   = row / Z;
    int h   = c >> 6;
    int dd  = c & 63;
    vt[(((size_t)b * H + h) * 64 + dd) * Z + z] = kv[(size_t)row * (2*F) + F + c];
}

// ==========================================================================
// Flash attention (causal, diagonal masked, mask = subtract 1e12, no 1/sqrt(d))
// Q:[B,S,F] bf16, KV:[B,Z,2F] bf16 (key = cols [0,F)), Vt:[(B*H*64), Z] bf16
// O:[B,S,F] bf16.  Grid: (S/128, H, B), 256 threads; one wave = 16 query rows.
// Software pipelined with a 2x ping-pong unroll: two K-fragment register sets
// alternate roles, so there is no loop-carried register rotation (no movs).
// ==========================================================================
__global__ __launch_bounds__(256)
void attn_kernel(const bf16* __restrict__ Q, const bf16* __restrict__ KV,
                 const bf16* __restrict__ Vt, bf16* __restrict__ O,
                 int S, int Z, int F, int H)
{
    constexpr int D = 64;
    const int lane  = threadIdx.x & 31;
    const int wid   = threadIdx.x >> 5;
    const int b     = blockIdx.z;
    const int h     = blockIdx.y;
    const int sBase = blockIdx.x * 128 + wid * 16;
    const int F2    = 2 * F;

    __shared__ __align__(16) bf16 Plds[8][16][40];   // per-wave P tile (16x32, padded)

    const int r     = lane & 15;
    const int half  = lane >> 4;
    const int halfA = half * 8;
    const int halfB = half * 16;

    // Q fragments (16 x 64 over two K=32 steps)
    v16bf qf[2];
    {
        const bf16* qrow = Q + ((size_t)b * S + (sBase + r)) * F + h * D;
        #pragma unroll
        for (int ks = 0; ks < 2; ++ks)
            qf[ks] = frag16(*(const v8bf*)(qrow + ks*32 + halfB),
                            *(const v8bf*)(qrow + ks*32 + halfB + 8));
    }

    float rmax[8], rsum[8];
    #pragma unroll
    for (int j = 0; j < 8; ++j) { rmax[j] = -1e30f; rsum[j] = 0.0f; }
    v8f acc[4] = {};

    const bf16* Kbase = KV + (size_t)b * Z * F2 + h * D;          // row z, stride 2F
    const bf16* Vbase = Vt + (size_t)((b * H + h) * D) * Z;       // row d, stride Z

    // K fragments: [zsub*2 + ks]
    auto loadKtile = [&](int zt, v16bf out[4]) {
        #pragma unroll
        for (int ks = 0; ks < 2; ++ks) {
            out[0*2 + ks] = *(const v16bf*)(Kbase + (size_t)(zt + r)      * F2 + ks*32 + halfB);
            out[1*2 + ks] = *(const v16bf*)(Kbase + (size_t)(zt + 16 + r) * F2 + ks*32 + halfB);
        }
    };

    // one 32-wide z-tile: consumes kcur, prefetches knext for zt+32
    auto ztile = [&](int zt, v16bf kcur[4], v16bf knext[4], bool loadNext) {
        // ---- V fragments early: independent of softmax, hide under WMMAs ----
        v16bf vbf[4];
        #pragma unroll
        for (int t = 0; t < 4; ++t)
            vbf[t] = *(const v16bf*)(Vbase + (size_t)(t*16 + r) * Z + zt + halfB);

        // ---- logits: 16(s) x 32(z) via 4 WMMAs (all operands resident) ----
        v8f s0 = {}, s1 = {};
        s0 = wmma_bf16(qf[0], kcur[0], s0);
        s1 = wmma_bf16(qf[0], kcur[2], s1);
        s0 = wmma_bf16(qf[1], kcur[1], s0);
        s1 = wmma_bf16(qf[1], kcur[3], s1);

        // ---- prefetch next z-tile's K fragments (overlaps softmax) ----
        if (loadNext) loadKtile(zt + 32, knext);

        // ---- mask + online softmax (rows live in 16-lane groups) ----
        const int z0 = zt + r, z1 = zt + 16 + r;
        #pragma unroll
        for (int j = 0; j < 8; ++j) {
            const int srow = sBase + half*8 + j;
            float l0 = s0[j] - ((z0 >= srow) ? 1e12f : 0.0f);
            float l1 = s1[j] - ((z1 >= srow) ? 1e12f : 0.0f);
            float mx = fmaxf(l0, l1);
            #pragma unroll
            for (int off = 8; off >= 1; off >>= 1) mx = fmaxf(mx, __shfl_xor(mx, off));
            float mnew  = fmaxf(rmax[j], mx);
            float scale = __expf(rmax[j] - mnew);
            rmax[j] = mnew;
            float p0 = __expf(l0 - mnew);
            float p1 = __expf(l1 - mnew);
            float ps = p0 + p1;
            #pragma unroll
            for (int off = 8; off >= 1; off >>= 1) ps += __shfl_xor(ps, off);
            rsum[j] = rsum[j] * scale + ps;
            #pragma unroll
            for (int t = 0; t < 4; ++t) acc[t][j] *= scale;
            Plds[wid][half*8 + j][r]      = to_bf(p0);
            Plds[wid][half*8 + j][16 + r] = to_bf(p1);
        }

        // ---- re-layout P (C-layout -> A-fragment) via wave-local LDS ----
        v16bf pf;
        {
            const bf16* pp = &Plds[wid][r][0];
            pf = frag16(*(const v8bf*)(pp + halfB), *(const v8bf*)(pp + halfB + 8));
        }

        // ---- O += P @ V  (4 d-subtiles, V already resident) ----
        #pragma unroll
        for (int t = 0; t < 4; ++t)
            acc[t] = wmma_bf16(pf, vbf[t], acc[t]);
    };

    v16bf kA[4], kB[4];
    loadKtile(0, kA);

    // Z/32 is even (Z=2048): ping-pong unroll by 2, no register rotation.
    for (int zt = 0; zt < Z; zt += 64) {
        ztile(zt,      kA, kB, true);             // zt+32 < Z always here
        ztile(zt + 32, kB, kA, (zt + 64) < Z);
    }

    // ---- normalize and store ----
    #pragma unroll
    for (int j = 0; j < 8; ++j) {
        const int srow = sBase + half*8 + j;
        const float inv = 1.0f / rsum[j];
        bf16* orow = O + ((size_t)b * S + srow) * F + h * D;
        #pragma unroll
        for (int t = 0; t < 4; ++t)
            orow[t*16 + r] = to_bf(acc[t][j] * inv);
    }
}

// ==========================================================================
extern "C" void kernel_launch(void* const* d_in, const int* in_sizes, int n_in,
                              void* d_out, int out_size, void* d_ws, size_t ws_size,
                              hipStream_t stream)
{
    (void)in_sizes; (void)n_in; (void)out_size; (void)ws_size;
    const float* attend_from = (const float*)d_in[0];
    const float* attend_to   = (const float*)d_in[1];
    const float* w_q   = (const float*)d_in[2];
    const float* b_q   = (const float*)d_in[3];
    const float* w_kv  = (const float*)d_in[4];
    const float* b_kv  = (const float*)d_in[5];
    const float* w_out = (const float*)d_in[6];
    const float* b_out = (const float*)d_in[7];
    float* out = (float*)d_out;

    const int B = 2, S = 2048, F = 1024, H = 16, Z = 2048;
    const int M = B * S;   // 4096 rows for all projections

    char* ws = (char*)d_ws;
    bf16* Qbf  = (bf16*)ws;  ws += (size_t)M * F * sizeof(bf16);        // 8 MB
    bf16* KVbf = (bf16*)ws;  ws += (size_t)M * 2 * F * sizeof(bf16);    // 16 MB
    bf16* Vtb  = (bf16*)ws;  ws += (size_t)M * F * sizeof(bf16);        // 8 MB
    bf16* Obf  = (bf16*)ws;                                             // 8 MB

    dim3 blk(256);

    // 1) Q projection
    gemm_bias_wmma<float, bf16><<<dim3(F / GM_BN, M / GM_BM), blk, 0, stream>>>(
        attend_from, w_q, b_q, Qbf, M, F, F);

    // 2) KV projection
    gemm_bias_wmma<float, bf16><<<dim3((2 * F) / GM_BN, M / GM_BM), blk, 0, stream>>>(
        attend_to, w_kv, b_kv, KVbf, M, 2 * F, F);

    // 3) V transpose -> Vt[b,h,d,Z]
    {
        int n = B * Z * F;
        v_transpose_kernel<<<n / 256, blk, 0, stream>>>(KVbf, Vtb, Z, F, H);
    }

    // 4) attention
    attn_kernel<<<dim3(S / 128, H, B), blk, 0, stream>>>(Qbf, KVbf, Vtb, Obf, S, Z, F, H);

    // 5) output projection (f32 out)
    gemm_bias_wmma<bf16, float><<<dim3(F / GM_BN, M / GM_BM), blk, 0, stream>>>(
        Obf, w_out, b_out, out, M, F, F);
}
